// OneStepGRU_28750511079433
// MI455X (gfx1250) — matrix operand order repaired and checked by reading, not compiled, not run on hardware
//
#include <hip/hip_runtime.h>

typedef __attribute__((ext_vector_type(16))) _Float16 v16h;
typedef __attribute__((ext_vector_type(8)))  _Float16 v8h;
typedef __attribute__((ext_vector_type(4)))  _Float16 v4h;
typedef __attribute__((ext_vector_type(8)))  float    v8f;

#define DIM    512
#define BM     32            // rows per workgroup
#define NW     8             // waves per workgroup (wave32)
#define WCOLS  64            // columns per wave
#define LDH    (DIM + 8)     // f16 LDS row stride (halfs): 1040B -> +4 banks/row
#define LDF    (DIM + 4)     // f32 LDS row stride (floats): 2064B -> +4 banks/row

union F16Frag { v16h v; float4 q[2]; };

__device__ __forceinline__ v8f wmma_f16(v16h a, v16h b, v8f c) {
  return __builtin_amdgcn_wmma_f32_16x16x32_f16(false, a, false, b,
                                                (short)0, c, false, false);
}

// A fragment (16x32 f16, M x K) from LDS, per ISA 7.12.2 layout:
// lane<16: row=lane,    K = {k0..k0+7, k0+16..k0+23}
// lane>=16: row=lane-16, K = {k0+8..k0+15, k0+24..k0+31}
__device__ __forceinline__ v16h load_a(const _Float16* p, int m0, int k0) {
  const int lane = threadIdx.x & 31;
  const int half = lane >> 4;
  const _Float16* base = p + (m0 + (lane & 15)) * LDH + k0 + half * 8;
  F16Frag a;
  a.q[0] = *(const float4*)(base);
  a.q[1] = *(const float4*)(base + 16);
  return a.v;
}

// B fragment (32x16 f16, K x N) from row-major f16 W[DIM][DIM]: B[k][n] = W[n][k].
// lane<16: col=n0+lane, K = {k0..k0+7, k0+16..k0+23} ; lane>=16: col=n0+lane-16, K = {k0+8..k0+15, k0+24..k0+31}
__device__ __forceinline__ v16h load_b(const _Float16* __restrict__ W, int n0, int k0) {
  const int lane = threadIdx.x & 31;
  const _Float16* base = W + (n0 + (lane & 15)) * DIM + k0 + (lane >> 4) * 8;
  F16Frag b;
  b.q[0] = *(const float4*)(base);
  b.q[1] = *(const float4*)(base + 16);
  return b.v;
}

__global__ void f32_to_f16_kernel(const float* __restrict__ s,
                                  _Float16* __restrict__ d, int n) {
  int i = (blockIdx.x * blockDim.x + threadIdx.x) * 4;
  if (i + 3 < n) {
    float4 v = *(const float4*)(s + i);
    v4h p = { (_Float16)v.x, (_Float16)v.y, (_Float16)v.z, (_Float16)v.w };
    *(v4h*)(d + i) = p;
  }
}

__global__ __launch_bounds__(256, 1)
void gru_fused_kernel(const float* __restrict__ x, const float* __restrict__ h,
                      const _Float16* __restrict__ Wh,   // [6][512][512]: rx,rh,zx,zh,nx,nh
                      const float* __restrict__ b_rx, const float* __restrict__ b_rh,
                      const float* __restrict__ b_zx, const float* __restrict__ b_zh,
                      const float* __restrict__ b_nx, const float* __restrict__ b_nh,
                      float* __restrict__ out) {
  __shared__ __align__(16) _Float16 sX [BM * LDH];
  __shared__ __align__(16) _Float16 sH [BM * LDH];
  __shared__ __align__(16) _Float16 sRH[BM * LDH];
  __shared__ __align__(16) _Float16 sZ [BM * LDH];
  __shared__ __align__(16) float    sNX[BM * LDF];

  const int tid   = threadIdx.x;
  const int lane  = tid & 31;
  const int wave  = tid >> 5;
  const int lhalf = lane >> 4;
  const int nn    = lane & 15;
  const long row0 = (long)blockIdx.x * BM;
  const int n0w   = wave * WCOLS;

  const _Float16* Wrx = Wh + 0 * DIM * DIM;
  const _Float16* Wrh = Wh + 1 * DIM * DIM;
  const _Float16* Wzx = Wh + 2 * DIM * DIM;
  const _Float16* Wzh = Wh + 3 * DIM * DIM;
  const _Float16* Wnx = Wh + 4 * DIM * DIM;
  const _Float16* Wnh = Wh + 5 * DIM * DIM;

  // ---- stage x,h stripe (32x512 f32 -> f16 LDS), fully coalesced b128 loads ----
  #pragma unroll
  for (int i = 0; i < 16; ++i) {
    int idx = tid + 256 * i;        // 0..4095
    int r   = idx >> 7;             // 0..31
    int c4  = idx & 127;            // float4 index within row
    float4 vx = *(const float4*)(x + (row0 + r) * DIM + 4 * c4);
    float4 vh = *(const float4*)(h + (row0 + r) * DIM + 4 * c4);
    v4h px = { (_Float16)vx.x, (_Float16)vx.y, (_Float16)vx.z, (_Float16)vx.w };
    v4h ph = { (_Float16)vh.x, (_Float16)vh.y, (_Float16)vh.z, (_Float16)vh.w };
    *(v4h*)&sX[r * LDH + 4 * c4] = px;
    *(v4h*)&sH[r * LDH + 4 * c4] = ph;
  }
  __syncthreads();

  // ================= pass 1: r = sigmoid(x@Wrx.T + h@Wrh.T + b) ; RH = r*h ====
  {
    v8f acc[2][4] = {};
    for (int k0 = 0; k0 < DIM; k0 += 32) {
      v16h ax0 = load_a(sX, 0, k0),  ax1 = load_a(sX, 16, k0);
      v16h ah0 = load_a(sH, 0, k0),  ah1 = load_a(sH, 16, k0);
      #pragma unroll
      for (int t = 0; t < 4; ++t) {
        v16h bx = load_b(Wrx, n0w + 16 * t, k0);
        v16h bh = load_b(Wrh, n0w + 16 * t, k0);
        acc[0][t] = wmma_f16(ax0, bx, acc[0][t]);
        acc[1][t] = wmma_f16(ax1, bx, acc[1][t]);
        acc[0][t] = wmma_f16(ah0, bh, acc[0][t]);
        acc[1][t] = wmma_f16(ah1, bh, acc[1][t]);
      }
    }
    #pragma unroll
    for (int rt = 0; rt < 2; ++rt)
      #pragma unroll
      for (int t = 0; t < 4; ++t) {
        int col = n0w + 16 * t + nn;
        float bb = b_rx[col] + b_rh[col];
        #pragma unroll
        for (int v = 0; v < 8; ++v) {
          int row = rt * 16 + v + lhalf * 8;
          float pre = acc[rt][t][v] + bb;
          float rr  = 1.0f / (1.0f + __expf(-pre));
          float hv  = (float)sH[row * LDH + col];
          sRH[row * LDH + col] = (_Float16)(rr * hv);
        }
      }
  }

  // ================= pass 2: z = sigmoid(x@Wzx.T + h@Wzh.T + b) ===============
  {
    v8f acc[2][4] = {};
    for (int k0 = 0; k0 < DIM; k0 += 32) {
      v16h ax0 = load_a(sX, 0, k0),  ax1 = load_a(sX, 16, k0);
      v16h ah0 = load_a(sH, 0, k0),  ah1 = load_a(sH, 16, k0);
      #pragma unroll
      for (int t = 0; t < 4; ++t) {
        v16h bx = load_b(Wzx, n0w + 16 * t, k0);
        v16h bh = load_b(Wzh, n0w + 16 * t, k0);
        acc[0][t] = wmma_f16(ax0, bx, acc[0][t]);
        acc[1][t] = wmma_f16(ax1, bx, acc[1][t]);
        acc[0][t] = wmma_f16(ah0, bh, acc[0][t]);
        acc[1][t] = wmma_f16(ah1, bh, acc[1][t]);
      }
    }
    #pragma unroll
    for (int rt = 0; rt < 2; ++rt)
      #pragma unroll
      for (int t = 0; t < 4; ++t) {
        int col = n0w + 16 * t + nn;
        float bb = b_zx[col] + b_zh[col];
        #pragma unroll
        for (int v = 0; v < 8; ++v) {
          int row = rt * 16 + v + lhalf * 8;
          float pre = acc[rt][t][v] + bb;
          sZ[row * LDH + col] = (_Float16)(1.0f / (1.0f + __expf(-pre)));
        }
      }
  }

  // ================= pass 3: NX = x@Wnx.T + b_nx (kept f32) ===================
  {
    v8f acc[2][4] = {};
    for (int k0 = 0; k0 < DIM; k0 += 32) {
      v16h ax0 = load_a(sX, 0, k0),  ax1 = load_a(sX, 16, k0);
      #pragma unroll
      for (int t = 0; t < 4; ++t) {
        v16h bx = load_b(Wnx, n0w + 16 * t, k0);
        acc[0][t] = wmma_f16(ax0, bx, acc[0][t]);
        acc[1][t] = wmma_f16(ax1, bx, acc[1][t]);
      }
    }
    #pragma unroll
    for (int rt = 0; rt < 2; ++rt)
      #pragma unroll
      for (int t = 0; t < 4; ++t) {
        int col = n0w + 16 * t + nn;
        float bb = b_nx[col];
        #pragma unroll
        for (int v = 0; v < 8; ++v) {
          int row = rt * 16 + v + lhalf * 8;
          sNX[row * LDF + col] = acc[rt][t][v] + bb;
        }
      }
  }

  __syncthreads();   // RH is read across all waves' columns below

  // ============ pass 4: NH = (r*h)@Wnh.T ; out = (1-z)*tanh(.) + z*h ==========
  {
    v8f acc[2][4] = {};
    for (int k0 = 0; k0 < DIM; k0 += 32) {
      v16h a0 = load_a(sRH, 0, k0),  a1 = load_a(sRH, 16, k0);
      #pragma unroll
      for (int t = 0; t < 4; ++t) {
        v16h bh = load_b(Wnh, n0w + 16 * t, k0);
        acc[0][t] = wmma_f16(a0, bh, acc[0][t]);
        acc[1][t] = wmma_f16(a1, bh, acc[1][t]);
      }
    }
    #pragma unroll
    for (int rt = 0; rt < 2; ++rt)
      #pragma unroll
      for (int t = 0; t < 4; ++t) {
        int col = n0w + 16 * t + nn;
        float bb = b_nh[col];
        #pragma unroll
        for (int v = 0; v < 8; ++v) {
          int row = rt * 16 + v + lhalf * 8;
          float nv = tanhf(sNX[row * LDF + col] + acc[rt][t][v] + bb);
          float zv = (float)sZ[row * LDH + col];
          float hv = (float)sH[row * LDH + col];
          out[(row0 + row) * DIM + col] = (1.0f - zv) * nv + zv * hv;
        }
      }
  }
}

extern "C" void kernel_launch(void* const* d_in, const int* in_sizes, int n_in,
                              void* d_out, int out_size, void* d_ws, size_t ws_size,
                              hipStream_t stream) {
  const float* x    = (const float*)d_in[0];
  const float* h    = (const float*)d_in[1];
  // setup_inputs order: x,h, W_rx,b_rx, W_rh,b_rh, W_zx,b_zx, W_zh,b_zh, W_nx,b_nx, W_nh,b_nh
  const float* Wsrc[6] = { (const float*)d_in[2],  (const float*)d_in[4],
                           (const float*)d_in[6],  (const float*)d_in[8],
                           (const float*)d_in[10], (const float*)d_in[12] };
  const float* b_rx = (const float*)d_in[3];
  const float* b_rh = (const float*)d_in[5];
  const float* b_zx = (const float*)d_in[7];
  const float* b_zh = (const float*)d_in[9];
  const float* b_nx = (const float*)d_in[11];
  const float* b_nh = (const float*)d_in[13];

  _Float16* Wh = (_Float16*)d_ws;            // 6*512*512 f16 = 3 MB (L2-resident)
  const int welts = DIM * DIM;
  for (int i = 0; i < 6; ++i)
    f32_to_f16_kernel<<<welts / (256 * 4), 256, 0, stream>>>(Wsrc[i], Wh + (size_t)i * welts, welts);

  const int nrows = in_sizes[0] / DIM;       // 131072
  gru_fused_kernel<<<nrows / BM, 256, 0, stream>>>(
      x, h, Wh, b_rx, b_rh, b_zx, b_zh, b_nx, b_nh, (float*)d_out);
}
